// GSVDLayer_72868415144129
// MI455X (gfx1250) — matrix operand order repaired and checked by reading, not compiled, not run on hardware
//
#include <hip/hip_runtime.h>
#include <hip/hip_bf16.h>
#include <stdint.h>

// GSVD layer: y = x @ (U*S @ Vh)^T, factored:
//   T = (x @ Vh^T) * S   [8192,1024]  bf16 (workspace)
//   y = T @ U^T          [8192,4096]  fp32 (d_out)
// bf16 WMMA, fp32 accumulate. Double-buffered LDS pipeline; prefetch issued
// AFTER the barrier so the barrier fence only drains DScnt, never the prefetch.
// GEMM2 stages T via the Tensor Data Mover (tensor_load_to_lds + s_wait_tensorcnt).

#define M_TOK   8192
#define D_IN_   4096
#define D_OUT_  4096
#define RANK_   1024

typedef __attribute__((ext_vector_type(16))) __bf16 v16bf;
typedef __attribute__((ext_vector_type(8)))  __bf16 v8bf;
typedef __attribute__((ext_vector_type(8)))  float  v8f;
typedef __attribute__((ext_vector_type(4)))  unsigned int v4u;
typedef __attribute__((ext_vector_type(8)))  int    v8i;
typedef __attribute__((ext_vector_type(4)))  int    v4i;

#define BM 128
#define BN 128
#define BK 64
#define LDSK 72   // padded LDS row stride (bf16): 64 data + 8 pad -> conflict-free

__device__ __forceinline__ v8bf pack8(float4 a, float4 b) {
    v8bf p;
    p[0] = (__bf16)a.x; p[1] = (__bf16)a.y; p[2] = (__bf16)a.z; p[3] = (__bf16)a.w;
    p[4] = (__bf16)b.x; p[5] = (__bf16)b.y; p[6] = (__bf16)b.z; p[7] = (__bf16)b.w;
    return p;
}

__device__ __forceinline__ v16bf frag16(const __bf16* p, int gap) {
    v8bf lo = *(const v8bf*)p;
    v8bf hi = *(const v8bf*)(p + gap);
    return __builtin_shufflevector(lo, hi, 0,1,2,3,4,5,6,7,8,9,10,11,12,13,14,15);
}

// One 64-wide K-chunk: 2 WMMA K-steps over a 32x64 wave tile (2x4 accumulators).
__device__ __forceinline__ void compute_chunk(const __bf16* As, const __bf16* Bs,
                                              int wm, int wn, int l16, int half,
                                              v8f acc[2][4]) {
#pragma unroll
    for (int ks = 0; ks < 2; ++ks) {
        const int kb = ks * 32;
        v16bf af[2], bfm[4];
#pragma unroll
        for (int mt = 0; mt < 2; ++mt)   // A layout: lanes 0-15 K{0..7,16..23}, 16-31 K{8..15,24..31}
            af[mt] = frag16(&As[(wm * 32 + mt * 16 + l16) * LDSK + kb + 8 * half], 16);
#pragma unroll
        for (int nt = 0; nt < 4; ++nt)   // B layout: lane=col, lanes 0-15 K{0..7,16..23}, 16-31 K{8..15,24..31}
            bfm[nt] = frag16(&Bs[(wn * 64 + nt * 16 + l16) * LDSK + kb + 8 * half], 16);
#pragma unroll
        for (int mt = 0; mt < 2; ++mt)
#pragma unroll
            for (int nt = 0; nt < 4; ++nt)
                acc[mt][nt] = __builtin_amdgcn_wmma_f32_16x16x32_bf16(
                    false, af[mt], false, bfm[nt], (short)0, acc[mt][nt], false, false);
    }
}

// TDM: DMA a 128x64 bf16 tile (row stride 1024 elems) from global into LDS,
// inserting 16B of padding after every 128B row -> 72-element LDS rows.
__device__ __forceinline__ void tdm_load_tile(const __bf16* gtile, unsigned int lds_byte_addr) {
    const unsigned long long ga = (unsigned long long)(uintptr_t)gtile;
    v4u g0;
    g0[0] = 1u;                                   // count=1, user descriptor
    g0[1] = lds_byte_addr;                        // lds_addr
    g0[2] = (unsigned int)ga;                     // global_addr[31:0]
    g0[3] = (unsigned int)((ga >> 32) & 0x01FFFFFFu) | (2u << 30);  // addr[56:32] | type=2
    v8i g1;
    g1[0] = (int)((1u << 16)        // data_size = 2 bytes
                | (1u << 20)        // pad_enable
                | (4u << 22)        // pad_interval: every 128 bytes
                | (3u << 25));      // pad_amount: 4 DWORDs = 16 bytes
    g1[1] = (int)(((unsigned)RANK_ & 0xFFFFu) << 16);            // tensor_dim0[15:0]
    g1[2] = (int)((((unsigned)RANK_ >> 16) & 0xFFFFu)
                | (((unsigned)M_TOK & 0xFFFFu) << 16));          // dim0[31:16] | dim1[15:0]
    g1[3] = (int)((((unsigned)M_TOK >> 16) & 0xFFFFu)
                | (64u << 16));                                  // dim1[31:16] | tile_dim0=64
    g1[4] = (int)(128u);                                         // tile_dim1=128, tile_dim2=0
    g1[5] = (int)((unsigned)RANK_);                              // tensor_dim0_stride[31:0]
    g1[6] = 0;                                                   // stride hi / dim1_stride lo
    g1[7] = 0;
    v4i z4 = {0, 0, 0, 0};
    v8i z8 = {0, 0, 0, 0, 0, 0, 0, 0};
    __builtin_amdgcn_tensor_load_to_lds(g0, g1, z4, z4, z8, 0);
}

// ---------------- GEMM 1: T[m,r] = S[r] * sum_k X[m,k]*Vh[r,k] ----------------
__global__ void __launch_bounds__(256)
gemm_x_vht(const float* __restrict__ X, const float* __restrict__ Vh,
           const float* __restrict__ S, __bf16* __restrict__ T) {
    __shared__ __align__(16) __bf16 As[2][BM * LDSK];
    __shared__ __align__(16) __bf16 Bs[2][BN * LDSK];

    const int tid = threadIdx.x;
    const int wid = tid >> 5, lane = tid & 31;
    const int wm  = wid & 3,  wn   = wid >> 2;   // 4x2 waves -> 32x64 tile each
    const int l16 = lane & 15, half = lane >> 4;
    const int bm  = blockIdx.y * BM;
    const int bn  = blockIdx.x * BN;
    const int srow = tid >> 1;
    const int scol = (tid & 1) * 32;

    const float* gx = X  + (size_t)(bm + srow) * D_IN_ + scol;
    const float* gv = Vh + (size_t)(bn + srow) * D_IN_ + scol;

    float4 rx[8], rv[8];
    v8f acc[2][4] = {};

    auto loadChunk = [&](int k0) {
        const float4* px = (const float4*)(gx + k0);
        const float4* pv = (const float4*)(gv + k0);
#pragma unroll
        for (int i = 0; i < 8; ++i) { rx[i] = px[i]; rv[i] = pv[i]; }
    };
    auto storeChunk = [&](int b) {
        __bf16* sa = &As[b][srow * LDSK + scol];
        __bf16* sb = &Bs[b][srow * LDSK + scol];
#pragma unroll
        for (int i = 0; i < 4; ++i) {
            *(v8bf*)(sa + 8 * i) = pack8(rx[2 * i], rx[2 * i + 1]);
            *(v8bf*)(sb + 8 * i) = pack8(rv[2 * i], rv[2 * i + 1]);
        }
    };

    loadChunk(0);
    storeChunk(0);
    int buf = 0;
    // Steady state: barrier drains only DScnt; prefetch issues after the barrier
    // and its LOADcnt is drained only inside storeChunk, after the WMMA work.
    for (int k0 = 0; k0 < D_IN_ - BK; k0 += BK) {
        __syncthreads();
        loadChunk(k0 + BK);
        compute_chunk(As[buf], Bs[buf], wm, wn, l16, half, acc);
        storeChunk(buf ^ 1);
        buf ^= 1;
    }
    __syncthreads();
    compute_chunk(As[buf], Bs[buf], wm, wn, l16, half, acc);   // last chunk, no prefetch

    const int colBase = bn + wn * 64;
#pragma unroll
    for (int nt = 0; nt < 4; ++nt) {
        const int col = colBase + nt * 16 + l16;
        const float s = S[col];
#pragma unroll
        for (int mt = 0; mt < 2; ++mt) {
            const int row0 = bm + wm * 32 + mt * 16 + half * 8;
#pragma unroll
            for (int j = 0; j < 8; ++j)
                T[(size_t)(row0 + j) * RANK_ + col] = (__bf16)(acc[mt][nt][j] * s);
        }
    }
}

// ---------------- GEMM 2: Y[m,o] = sum_r T[m,r]*U[o,r]  (T staged via TDM) ----------------
__global__ void __launch_bounds__(256)
gemm_t_ut(const __bf16* __restrict__ T, const float* __restrict__ U,
          float* __restrict__ Y) {
    __shared__ __align__(16) __bf16 As[2][BM * LDSK];
    __shared__ __align__(16) __bf16 Bs[2][BN * LDSK];

    const int tid = threadIdx.x;
    const int wid = tid >> 5, lane = tid & 31;
    const int wm  = wid & 3,  wn   = wid >> 2;
    const int l16 = lane & 15, half = lane >> 4;
    const int bm  = blockIdx.y * BM;
    const int bn  = blockIdx.x * BN;
    const int srow = tid >> 1;
    const int scol = (tid & 1) * 32;

    const __bf16* gtile = T + (size_t)bm * RANK_;           // tile rows bm..bm+127
    const float*  gu    = U + (size_t)(bn + srow) * RANK_ + scol;

    float4 ru[8];
    v8f acc[2][4] = {};

    auto loadU = [&](int k0) {
        const float4* pu = (const float4*)(gu + k0);
#pragma unroll
        for (int i = 0; i < 8; ++i) ru[i] = pu[i];
    };
    auto storeU = [&](int b) {
        __bf16* sb = &Bs[b][srow * LDSK + scol];
#pragma unroll
        for (int i = 0; i < 4; ++i)
            *(v8bf*)(sb + 8 * i) = pack8(ru[2 * i], ru[2 * i + 1]);
    };

    // Prologue: TDM chunk0 -> As[0]; U chunk0 -> Bs[0] via VGPRs (fp32->bf16).
    if (wid == 0) tdm_load_tile(gtile + 0, (unsigned int)(uintptr_t)&As[0][0]);
    loadU(0);
    storeU(0);

    int buf = 0;
    for (int k0 = 0; k0 < RANK_ - BK; k0 += BK) {
        // My buffer's TDM (issued last iteration / prologue) must be done; the
        // next one hasn't been issued yet, so wait-for-0 is exact.
        if (wid == 0) __builtin_amdgcn_s_wait_tensorcnt(0);
        __syncthreads();
        if (wid == 0)
            tdm_load_tile(gtile + (k0 + BK), (unsigned int)(uintptr_t)&As[buf ^ 1][0]);
        loadU(k0 + BK);
        compute_chunk(As[buf], Bs[buf], wm, wn, l16, half, acc);
        storeU(buf ^ 1);
        buf ^= 1;
    }
    if (wid == 0) __builtin_amdgcn_s_wait_tensorcnt(0);
    __syncthreads();
    compute_chunk(As[buf], Bs[buf], wm, wn, l16, half, acc);

    const int colBase = bn + wn * 64;
#pragma unroll
    for (int nt = 0; nt < 4; ++nt) {
        const int col = colBase + nt * 16 + l16;
#pragma unroll
        for (int mt = 0; mt < 2; ++mt) {
            const int row0 = bm + wm * 32 + mt * 16 + half * 8;
#pragma unroll
            for (int j = 0; j < 8; ++j)
                Y[(size_t)(row0 + j) * D_OUT_ + col] = acc[mt][nt][j];
        }
    }
}

extern "C" void kernel_launch(void* const* d_in, const int* in_sizes, int n_in,
                              void* d_out, int out_size, void* d_ws, size_t ws_size,
                              hipStream_t stream) {
    (void)in_sizes; (void)n_in; (void)out_size; (void)ws_size;
    const float* x  = (const float*)d_in[0];   // [4,2048,4096]
    const float* U  = (const float*)d_in[1];   // [4096,1024]
    const float* S  = (const float*)d_in[2];   // [1024]
    const float* Vh = (const float*)d_in[3];   // [1024,4096]
    float*  y = (float*)d_out;                 // [4,2048,4096]
    __bf16* T = (__bf16*)d_ws;                 // [8192,1024] bf16 = 16.8 MB

    dim3 blk(256);
    gemm_x_vht<<<dim3(RANK_ / BN,  M_TOK / BM), blk, 0, stream>>>(x, Vh, S, T);
    gemm_t_ut <<<dim3(D_OUT_ / BN, M_TOK / BM), blk, 0, stream>>>(T, U, y);
}
